// BiLstmClassifier_3796751089853
// MI455X (gfx1250) — matrix operand order repaired and checked by reading, not compiled, not run on hardware
//
#include <hip/hip_runtime.h>
#include <hip/hip_bf16.h>
#include <math.h>

typedef __attribute__((ext_vector_type(16))) __bf16       v16bf;
typedef __attribute__((ext_vector_type(8)))  float        v8f;
typedef __attribute__((ext_vector_type(4)))  unsigned int u32x4;

union Frag { u32x4 q[2]; v16bf v; };

__device__ __forceinline__ unsigned short f2bf(float f) {
    unsigned int u = __float_as_uint(f);
    u += 0x7FFFu + ((u >> 16) & 1u);          // round-to-nearest-even
    return (unsigned short)(u >> 16);
}

#define BLK 512   // 16 wave32's per block
#define NWAVES (BLK / 32)

constexpr size_t lstm_smem(int D, int H, bool mean) {
    int DK = ((D + 31) / 32) * 32;
    int HK = ((H + 31) / 32) * 32;
    int KT = DK + HK;
    int KC = KT / 32;
    int G4 = 4 * H;
    int NT = G4 / 16;
    int GS = G4 + 4;
    size_t s = (size_t)NT * KC * 512 * 2   // packed B fragments (bf16)
             + (size_t)16 * KT * 2         // A panel [x_t | h] (bf16)
             + (size_t)G4 * 4              // bias
             + (size_t)16 * GS * 4         // pre-activation gates (f32)
             + (size_t)16 * H * 4;         // cell state c (f32)
    if (mean) s += (size_t)16 * H * 4;     // running mean accumulator
    return s;
}

// One bidirectional-LSTM direction per block. Block handles a 16-row batch
// tile for all T steps: fused GEMM gates = [x_t,h]*[Wih;Whh]^T + b via WMMA
// with register-resident weights, then elementwise LSTM cell update.
template<int D, int H, bool IN_F32, bool MEAN>
__global__ __launch_bounds__(BLK)
void lstm_layer(const float* __restrict__ xf,          // f32 input (layer 1)
                const unsigned short* __restrict__ xb, // bf16 input (layers 2,3)
                const float* __restrict__ Wih,         // (2,4H,D)
                const float* __restrict__ Whh,         // (2,4H,H)
                const float* __restrict__ bih,         // (2,4H)
                const float* __restrict__ bhh,         // (2,4H)
                unsigned short* __restrict__ outb,     // (B,T,2H) bf16
                float* __restrict__ sumout,            // (B,2H) f32 if MEAN
                int T)
{
    constexpr int DK = ((D + 31) / 32) * 32;
    constexpr int HK = ((H + 31) / 32) * 32;
    constexpr int KT = DK + HK;        // padded K of fused GEMM
    constexpr int KC = KT / 32;        // K-chunks of 32
    constexpr int G4 = 4 * H;
    constexpr int NT = G4 / 16;        // 16-col N tiles
    constexpr int GS = G4 + 4;         // gates LDS row stride (floats)
    constexpr int TPW = (NT + NWAVES - 1) / NWAVES;  // N tiles per wave

    extern __shared__ char smem_raw[];
    unsigned short* Bpack = (unsigned short*)smem_raw;       // NT*KC*512
    unsigned short* Afull = Bpack + NT * KC * 512;           // 16*KT
    float* bias  = (float*)(Afull + 16 * KT);                // G4
    float* gates = bias + G4;                                // 16*GS
    float* cst   = gates + 16 * GS;                          // 16*H
    float* msum  = cst + 16 * H;                             // 16*H (MEAN only)

    const int tid = threadIdx.x;
    const int dir = (int)blockIdx.x & 1;
    const int b0  = ((int)blockIdx.x >> 1) * 16;

    const float* wih = Wih + (size_t)dir * G4 * D;
    const float* whh = Whh + (size_t)dir * G4 * H;

    // ---- one-time: pack [Wih;Whh]^T into per-lane WMMA B-fragment layout ----
    // fragment element e, lane ln: K = kc*32 + (ln>=16 ? 8:0) + (e<8 ? e : 16+e-8), N = nt*16 + (ln&15)
    for (int i = tid; i < NT * KC * 512; i += BLK) {
        int e  = i & 15;
        int ln = (i >> 4) & 31;
        int kc = (i >> 9) % KC;
        int nt = i / (KC * 512);
        int k  = kc * 32 + ((ln & 16) ? 8 : 0) + (e < 8 ? e : 16 + (e - 8));
        int n  = nt * 16 + (ln & 15);
        float v = 0.f;
        if (k < DK) { if (k < D) v = wih[n * D + k]; }
        else        { int kh = k - DK; if (kh < H) v = whh[n * H + kh]; }
        Bpack[i] = f2bf(v);
    }
    for (int i = tid; i < G4; i += BLK) bias[i] = bih[dir * G4 + i] + bhh[dir * G4 + i];
    for (int i = tid; i < 16 * KT; i += BLK) Afull[i] = 0;   // zero pads + h0 = 0
    for (int i = tid; i < 16 * H; i += BLK) { cst[i] = 0.f; if (MEAN) msum[i] = 0.f; }
    __syncthreads();

    const int lane = tid & 31;
    const int wv   = tid >> 5;
    const int mrow = lane & 15;
    const int half = (lane & 16) ? 8 : 0;   // A: K sub-offset; C: M sub-offset

    // ---- hoist weights + bias into registers (invariant over all T steps) ----
    Frag  breg[TPW][KC];
    float bvreg[TPW];
    #pragma unroll
    for (int ti = 0; ti < TPW; ++ti) {
        const int nt = wv + ti * NWAVES;
        bvreg[ti] = 0.f;
        if (nt < NT) {
            bvreg[ti] = bias[nt * 16 + mrow];
            const unsigned short* brow = Bpack + (size_t)nt * KC * 512 + lane * 16;
            #pragma unroll
            for (int kc = 0; kc < KC; ++kc) {
                breg[ti][kc].q[0] = *(const u32x4*)(brow + kc * 512);
                breg[ti][kc].q[1] = *(const u32x4*)(brow + kc * 512 + 8);
            }
        }
    }

    // ---- stage x_t into A panel (cols [0,D)) ----
    auto stage_x = [&](int t) {
        const int tt = dir ? (T - 1 - t) : t;
        if constexpr (IN_F32) {
            for (int i = tid; i < 16 * D; i += BLK) {
                int m = i / D, k = i - m * D;
                Afull[m * KT + k] = f2bf(xf[((size_t)(b0 + m) * T + tt) * D + k]);
            }
        } else {
            for (int i = tid; i < 16 * D; i += BLK) {
                int m = i / D, k = i - m * D;
                Afull[m * KT + k] = xb[((size_t)(b0 + m) * T + tt) * D + k];
            }
            if (t + 1 < T && tid < 16) {     // prefetch next step's tile row
                int tn = dir ? (T - 2 - t) : (t + 1);
                __builtin_prefetch(&xb[((size_t)(b0 + tid) * T + tn) * D], 0, 1);
            }
        }
    };

    stage_x(0);

    for (int t = 0; t < T; ++t) {
        __syncthreads();   // A panel (x_t | h_t) ready

        // ---- fused GEMM: gates = [x_t,h] * [Wih;Whh]^T + bias ----
        Frag areg[KC];
        {
            const unsigned short* arow = Afull + mrow * KT + half;
            #pragma unroll
            for (int kc = 0; kc < KC; ++kc) {
                areg[kc].q[0] = *(const u32x4*)(arow + kc * 32);
                areg[kc].q[1] = *(const u32x4*)(arow + kc * 32 + 16);
            }
        }
        #pragma unroll
        for (int ti = 0; ti < TPW; ++ti) {
            const int nt = wv + ti * NWAVES;
            if (nt < NT) {
                v8f acc;
                #pragma unroll
                for (int r = 0; r < 8; ++r) acc[r] = bvreg[ti];
                #pragma unroll
                for (int kc = 0; kc < KC; ++kc)
                    acc = __builtin_amdgcn_wmma_f32_16x16x32_bf16(
                              false, areg[kc].v, false, breg[ti][kc].v,
                              (short)0, acc, false, false);
                float* gp = gates + half * GS + nt * 16 + mrow;
                #pragma unroll
                for (int r = 0; r < 8; ++r) gp[r * GS] = acc[r];
            }
        }
        __syncthreads();   // gates ready

        // ---- elementwise LSTM cell (PyTorch gate order i,f,g,o) ----
        const int tt = dir ? (T - 1 - t) : t;
        for (int cell = tid; cell < 16 * H; cell += BLK) {
            int m = cell / H, j = cell - m * H;
            float gi = gates[m * GS + j];
            float gf = gates[m * GS + H + j];
            float gg = gates[m * GS + 2 * H + j];
            float go = gates[m * GS + 3 * H + j];
            float ig = 1.f / (1.f + __expf(-gi));
            float fg = 1.f / (1.f + __expf(-gf));
            float og = 1.f / (1.f + __expf(-go));
            float c  = fg * cst[cell] + ig * tanhf(gg);
            cst[cell] = c;
            float h = og * tanhf(c);
            Afull[m * KT + DK + j] = f2bf(h);   // h state lives inside A panel
            if constexpr (MEAN) msum[cell] += h;
            else outb[((size_t)(b0 + m) * T + tt) * (2 * H) + dir * H + j] = f2bf(h);
        }
        if (t + 1 < T) stage_x(t + 1);   // disjoint A-panel cols; same barrier interval
    }

    if constexpr (MEAN) {
        __syncthreads();
        for (int cell = tid; cell < 16 * H; cell += BLK) {
            int m = cell / H, j = cell - m * H;
            sumout[(b0 + m) * (2 * H) + dir * H + j] = msum[cell];
        }
    }
}

// mean (sum/T) + 32->64->16->20 MLP head; trivial work, one row per thread.
__global__ __launch_bounds__(256)
void head_kernel(const float* __restrict__ sum3,
                 const float* __restrict__ hW1, const float* __restrict__ hb1,
                 const float* __restrict__ hW2, const float* __restrict__ hb2,
                 const float* __restrict__ hW3, const float* __restrict__ hb3,
                 float* __restrict__ out, int B, float invT)
{
    int b = blockIdx.x * blockDim.x + threadIdx.x;
    if (b >= B) return;
    float f[32], h1[64], h2[16];
    for (int k = 0; k < 32; ++k) f[k] = sum3[b * 32 + k] * invT;
    for (int i = 0; i < 64; ++i) {
        float a = hb1[i];
        for (int k = 0; k < 32; ++k) a += f[k] * hW1[i * 32 + k];
        h1[i] = fmaxf(a, 0.f);
    }
    for (int i = 0; i < 16; ++i) {
        float a = hb2[i];
        for (int k = 0; k < 64; ++k) a += h1[k] * hW2[i * 64 + k];
        h2[i] = fmaxf(a, 0.f);
    }
    for (int i = 0; i < 20; ++i) {
        float a = hb3[i];
        for (int k = 0; k < 16; ++k) a += h2[k] * hW3[i * 16 + k];
        out[b * 20 + i] = a;
    }
}

extern "C" void kernel_launch(void* const* d_in, const int* in_sizes, int n_in,
                              void* d_out, int out_size, void* d_ws, size_t ws_size,
                              hipStream_t stream)
{
    const float* x    = (const float*)d_in[0];
    const float* Wih1 = (const float*)d_in[1];
    const float* Whh1 = (const float*)d_in[2];
    const float* bih1 = (const float*)d_in[3];
    const float* bhh1 = (const float*)d_in[4];
    const float* Wih2 = (const float*)d_in[5];
    const float* Whh2 = (const float*)d_in[6];
    const float* bih2 = (const float*)d_in[7];
    const float* bhh2 = (const float*)d_in[8];
    const float* Wih3 = (const float*)d_in[9];
    const float* Whh3 = (const float*)d_in[10];
    const float* bih3 = (const float*)d_in[11];
    const float* bhh3 = (const float*)d_in[12];
    const float* hW1  = (const float*)d_in[13];
    const float* hb1  = (const float*)d_in[14];
    const float* hW2  = (const float*)d_in[15];
    const float* hb2  = (const float*)d_in[16];
    const float* hW3  = (const float*)d_in[17];
    const float* hb3  = (const float*)d_in[18];

    const int B = 256, T = 2048;

    char* ws = (char*)d_ws;
    unsigned short* act1 = (unsigned short*)ws;                      // (B,T,256) bf16
    size_t o1 = (size_t)B * T * 256 * sizeof(unsigned short);
    unsigned short* act2 = (unsigned short*)(ws + o1);               // (B,T,128) bf16
    size_t o2 = o1 + (size_t)B * T * 128 * sizeof(unsigned short);
    float* sum3 = (float*)(ws + o2);                                 // (B,32) f32

    constexpr size_t s1 = lstm_smem(2, 128, false);   // ~207 KB
    constexpr size_t s2 = lstm_smem(256, 64, false);  // ~191 KB
    constexpr size_t s3 = lstm_smem(128, 16, true);   // ~32 KB

    hipFuncSetAttribute((const void*)lstm_layer<2, 128, true,  false>,
                        hipFuncAttributeMaxDynamicSharedMemorySize, (int)s1);
    hipFuncSetAttribute((const void*)lstm_layer<256, 64, false, false>,
                        hipFuncAttributeMaxDynamicSharedMemorySize, (int)s2);
    hipFuncSetAttribute((const void*)lstm_layer<128, 16, false, true>,
                        hipFuncAttributeMaxDynamicSharedMemorySize, (int)s3);

    dim3 grid((B / 16) * 2), blk(BLK);
    lstm_layer<2, 128, true, false><<<grid, blk, s1, stream>>>(
        x, nullptr, Wih1, Whh1, bih1, bhh1, act1, nullptr, T);
    lstm_layer<256, 64, false, false><<<grid, blk, s2, stream>>>(
        nullptr, act1, Wih2, Whh2, bih2, bhh2, act2, nullptr, T);
    lstm_layer<128, 16, false, true><<<grid, blk, s3, stream>>>(
        nullptr, act2, Wih3, Whh3, bih3, bhh3, nullptr, sum3, T);
    head_kernel<<<dim3(1), dim3(256), 0, stream>>>(
        sum3, hW1, hb1, hW2, hb2, hW3, hb3, (float*)d_out, B, 1.0f / (float)T);

    (void)in_sizes; (void)n_in; (void)out_size; (void)ws_size;
}
